// Decoder_47940424958353
// MI455X (gfx1250) — matrix operand order repaired and checked by reading, not compiled, not run on hardware
//
#include <hip/hip_runtime.h>
#include <hip/hip_bf16.h>
#include <math.h>

// ---------------- problem constants ----------------
#define B_    32
#define H_    16
#define W_    64
#define HW_   1024
#define L_    48
#define D_    512
#define N_    256
#define A_    512
#define K_    112
#define KRE_  9
#define NPIX  (B_*HW_)   // 32768

typedef __attribute__((ext_vector_type(16))) __bf16 v16bf;
typedef __attribute__((ext_vector_type(8)))  float  v8f;

__device__ __forceinline__ void split_bf(float x, __bf16& hi, __bf16& lo) {
    hi = (__bf16)x;
    lo = (__bf16)(x - (float)hi);
}

// Fragment index maps (CDNA5 16-bit WMMA layouts, ISA 7.12.2)
//  A 16x32: lanes0-15 row=lane: K=0..7 (v0-3), 16..23 (v4-7); lanes16-31 row=lane-16: +8
__device__ __forceinline__ int a_kidx(int i, int half) {
    const int j = i >> 1, h = i & 1;
    return ((j < 4) ? 0 : 16) + half * 8 + ((j & 3) << 1) + h;
}
//  B 32x16: lanes0-15 col=lane: K=0..7 (v0-3), 16..23 (v4-7); lanes16-31 col=lane-16: +8
__device__ __forceinline__ int b_kidx(int i, int half) {
    const int j = i >> 1, h = i & 1;
    return ((j < 4) ? 0 : 16) + half * 8 + ((j & 3) << 1) + h;
}

// ============ weight packer: f32 [K,N] -> fragment-major bf16 hi/lo records ============
// record (c = k-chunk, nt = n-tile, lane): 32 bf16 = [0..15]=hi frag, [16..31]=lo frag
__global__ void pack_w(const float* __restrict__ Wm, int ldw, int ntiles,
                       int total, __bf16* __restrict__ out) {
    const int rec = blockIdx.x * blockDim.x + threadIdx.x;
    if (rec >= total) return;
    const int lane = rec & 31;
    const int nt = (rec >> 5) % ntiles;
    const int c  = (rec >> 5) / ntiles;
    const int half = lane >> 4, lr = lane & 15;
    __bf16* dst = out + (size_t)rec * 32;
#pragma unroll
    for (int i = 0; i < 16; ++i) {
        const float wv = Wm[(size_t)(c * 32 + b_kidx(i, half)) * ldw + nt * 16 + lr];
        __bf16 hi, lo; split_bf(wv, hi, lo);
        dst[i] = hi; dst[16 + i] = lo;
    }
}

// ============ ctx packer: ctx_val [B,D,HW] -> A-fragment-major bf16 hi/lo ============
// record (mt = pixel tile, c = k-chunk over D, lane)
__global__ void pack_actx(const float* __restrict__ ctx, __bf16* __restrict__ out) {
    const int rec = blockIdx.x * blockDim.x + threadIdx.x;  // (NPIX/16)*16*32
    if (rec >= (NPIX / 16) * 16 * 32) return;
    const int lane = rec & 31;
    const int c  = (rec >> 5) & 15;
    const int mt = (rec >> 5) >> 4;
    const int half = lane >> 4, lr = lane & 15;
    const int p = mt * 16 + lr;          // A row = pixel
    const int b = p >> 10, hw = p & 1023;
    __bf16* dst = out + (size_t)rec * 32;
#pragma unroll
    for (int i = 0; i < 16; ++i) {
        const int d = c * 32 + a_kidx(i, half);
        const float xv = ctx[((size_t)b * D_ + d) * HW_ + hw];
        __bf16 hi, lo; split_bf(xv, hi, lo);
        dst[i] = hi; dst[16 + i] = lo;
    }
}

// ============ GEMM, both operands packed (big pixel GEMMs): Out = A@B + bias ============
// wave tile 16x64: 4 accumulators, 12 WMMAs per k-chunk, no LDS.
__global__ void gemm_pp(const __bf16* __restrict__ Ap, const __bf16* __restrict__ Bp,
                        int ntiles, int kchunks, const float* __restrict__ bias,
                        float* __restrict__ Out, int ldo) {
    const int lane = threadIdx.x;
    const int mt = blockIdx.y, nbase = blockIdx.x * 4;
    const int half = lane >> 4, lr = lane & 15;
    v8f acc[4] = {};
    for (int c = 0; c < kchunks; ++c) {
        const v16bf* ap = (const v16bf*)Ap + ((size_t)(mt * kchunks + c) * 32 + lane) * 2;
        const v16bf ahi = ap[0], alo = ap[1];
        if (c + 1 < kchunks) __builtin_prefetch((const void*)(ap + 64), 0, 1);
#pragma unroll
        for (int u = 0; u < 4; ++u) {
            const v16bf* bp = (const v16bf*)Bp + ((size_t)(c * ntiles + nbase + u) * 32 + lane) * 2;
            const v16bf bhi = bp[0], blo = bp[1];
            acc[u] = __builtin_amdgcn_wmma_f32_16x16x32_bf16(false, ahi, false, bhi, (short)0, acc[u], false, false);
            acc[u] = __builtin_amdgcn_wmma_f32_16x16x32_bf16(false, ahi, false, blo, (short)0, acc[u], false, false);
            acc[u] = __builtin_amdgcn_wmma_f32_16x16x32_bf16(false, alo, false, bhi, (short)0, acc[u], false, false);
        }
    }
#pragma unroll
    for (int u = 0; u < 4; ++u) {
        const int ncol = (nbase + u) * 16 + lr;
        const int m0 = mt * 16 + half * 8;
#pragma unroll
        for (int j = 0; j < 8; ++j) {
            float v = acc[u][j];
            if (bias) v += bias[ncol];
            Out[(size_t)(m0 + j) * ldo + ncol] = v;
        }
    }
}

// ============ GEMM, f32 X + packed B: Out = X@B + bias + Cin ============
// AMODE 0: X row-major [M,K] stride ldx (activations)
// AMODE 2: X = att_past [B,HW]; col t<121 -> 11x11 neighborhood tap (K padded to 128)
template<int AMODE>
__global__ void gemm_xp(const float* __restrict__ X, int ldx,
                        const __bf16* __restrict__ Bp, int ntiles, int kchunks,
                        const float* __restrict__ bias, const float* __restrict__ Cin,
                        float* __restrict__ Out, int ldo) {
    __shared__ float Xs[16 * 32];
    const int lane = threadIdx.x;
    const int mt = blockIdx.y, nbase = blockIdx.x * 4;
    const int half = lane >> 4, lr = lane & 15;
    v8f acc[4] = {};
    for (int c = 0; c < kchunks; ++c) {
        const int k0 = c << 5;
        if (AMODE == 0) {
#pragma unroll
            for (int i = lane; i < 16 * 8; i += 32) {
                const int r = i >> 3, c4 = (i & 7) << 2;
                const float4 v = *(const float4*)(X + (size_t)(mt * 16 + r) * ldx + k0 + c4);
                Xs[r*32+c4+0] = v.x; Xs[r*32+c4+1] = v.y; Xs[r*32+c4+2] = v.z; Xs[r*32+c4+3] = v.w;
            }
        } else {
#pragma unroll
            for (int r = 0; r < 16; ++r) {
                const int p = mt * 16 + r;
                const int b = p >> 10, hw = p & 1023, hh0 = hw >> 6, ww0 = hw & 63;
                const int t = k0 + lane;
                float v = 0.f;
                if (t < 121) {
                    const int hh = hh0 + t / 11 - 5, ww = ww0 + t % 11 - 5;
                    if (hh >= 0 && hh < H_ && ww >= 0 && ww < W_)
                        v = X[(size_t)b * HW_ + hh * W_ + ww];
                }
                Xs[r * 32 + lane] = v;
            }
        }
        __syncthreads();   // single-wave WG: S_NOP in hardware, orders LDS for compiler
        v16bf ahi, alo;
#pragma unroll
        for (int i = 0; i < 16; ++i) {
            const float xv = Xs[lr * 32 + a_kidx(i, half)];
            __bf16 xh, xl; split_bf(xv, xh, xl);
            ahi[i] = xh; alo[i] = xl;
        }
#pragma unroll
        for (int u = 0; u < 4; ++u) {
            const v16bf* bp = (const v16bf*)Bp + ((size_t)(c * ntiles + nbase + u) * 32 + lane) * 2;
            const v16bf bhi = bp[0], blo = bp[1];
            acc[u] = __builtin_amdgcn_wmma_f32_16x16x32_bf16(false, ahi, false, bhi, (short)0, acc[u], false, false);
            acc[u] = __builtin_amdgcn_wmma_f32_16x16x32_bf16(false, ahi, false, blo, (short)0, acc[u], false, false);
            acc[u] = __builtin_amdgcn_wmma_f32_16x16x32_bf16(false, alo, false, bhi, (short)0, acc[u], false, false);
        }
        __syncthreads();
    }
#pragma unroll
    for (int u = 0; u < 4; ++u) {
        const int ncol = (nbase + u) * 16 + lr;
        const int m0 = mt * 16 + half * 8;
#pragma unroll
        for (int j = 0; j < 8; ++j) {
            float v = acc[u][j];
            if (bias) v += bias[ncol];
            if (Cin)  v += Cin[(size_t)(m0 + j) * ldo + ncol];
            Out[(size_t)(m0 + j) * ldo + ncol] = v;
        }
    }
}

// ---------------- elementwise / reduction kernels ----------------
__global__ void init_kernel(const float* __restrict__ init_state,
                            float* ht, float* htr, float* attpast) {
    const int i = blockIdx.x * blockDim.x + threadIdx.x;
    if (i < B_ * N_) { ht[i] = init_state[i]; htr[i] = init_state[i]; }
    if (i < NPIX) attpast[i] = 0.f;
}

__global__ void w2_kernel(const float* __restrict__ convw, const float* __restrict__ fcw,
                          float* __restrict__ w2) {
    const int idx = blockIdx.x * blockDim.x + threadIdx.x;  // 128*512
    if (idx >= 128 * A_) return;
    const int t = idx >> 9, a = idx & 511;
    float s = 0.f;
    if (t < 121)
        for (int o = 0; o < 512; ++o) s += convw[o * 121 + t] * fcw[o * A_ + a];
    w2[idx] = s;
}

// maxout(128 pairs) -> [128x112] matvec -> softmax over 112, per pixel
__global__ void pix_soft(const float* __restrict__ f, const float* __restrict__ pw,
                         const float* __restrict__ pb, float* __restrict__ pf) {
    __shared__ float mx[128];
    __shared__ float red[128];
    const int p = blockIdx.x, tid = threadIdx.x;  // 128 threads
    mx[tid] = fmaxf(f[(size_t)p * 256 + 2 * tid], f[(size_t)p * 256 + 2 * tid + 1]);
    __syncthreads();
    float v = -INFINITY;
    if (tid < K_) {
        float s = pb[tid];
        for (int j = 0; j < 128; ++j) s += mx[j] * pw[j * K_ + tid];
        v = s;
    }
    red[tid] = v; __syncthreads();
    for (int k = 64; k; k >>= 1) { if (tid < k) red[tid] = fmaxf(red[tid], red[tid + k]); __syncthreads(); }
    const float vmax = red[0]; __syncthreads();
    const float ev = (tid < K_) ? expf(v - vmax) : 0.f;
    red[tid] = ev; __syncthreads();
    for (int k = 64; k; k >>= 1) { if (tid < k) red[tid] += red[tid + k]; __syncthreads(); }
    if (tid < K_) pf[(size_t)p * K_ + tid] = ev / red[0];
}

__global__ void gather_kernel(const float* __restrict__ emb_obj, const float* __restrict__ emb_rel,
                              const int* __restrict__ pys, const int* __restrict__ pres,
                              const int* __restrict__ cys, float* x0, float* cy) {
    const int idx = blockIdx.x * blockDim.x + threadIdx.x;  // 32*512
    if (idx < B_ * 512) {
        const int b = idx >> 9, c = idx & 511;
        x0[idx] = (c < 256) ? emb_obj[pys[b] * 256 + c] : emb_rel[pres[b] * 256 + (c - 256)];
    }
    if (idx < B_ * 256) {
        const int b = idx >> 8, c = idx & 255;
        cy[idx] = emb_obj[cys[b] * 256 + c];
    }
}

__global__ void gru_combine(const float* __restrict__ gi, const float* __restrict__ gh,
                            const float* __restrict__ h, const float* __restrict__ mask,
                            float* __restrict__ hout) {
    const int idx = blockIdx.x * blockDim.x + threadIdx.x;
    if (idx >= B_ * N_) return;
    const int b = idx >> 8, n = idx & 255;
    const float* gib = gi + b * 768;
    const float* ghb = gh + b * 768;
    const float r = 1.f / (1.f + expf(-(gib[n] + ghb[n])));
    const float z = 1.f / (1.f + expf(-(gib[256 + n] + ghb[256 + n])));
    const float nn = tanhf(gib[512 + n] + r * ghb[512 + n]);
    float out = (1.f - z) * nn + z * h[idx];
    if (mask) { const float m = mask[b]; out = m * out + (1.f - m) * h[idx]; }
    hout[idx] = out;
}

// s[p] = sum_a tanh(key[p,a] + q[b,a] (+ past[p,a])) * attw[a] + attb
__global__ void score_kernel(const float* __restrict__ key, const float* __restrict__ past,
                             const float* __restrict__ q, const float* __restrict__ attw,
                             const float* __restrict__ attb, float* __restrict__ s) {
    const int wid = (blockIdx.x * blockDim.x + threadIdx.x) >> 5;
    const int lane = threadIdx.x & 31;
    if (wid >= NPIX) return;
    const int b = wid >> 10;
    float sum = 0.f;
    for (int a = lane; a < A_; a += 32) {
        float v = key[(size_t)wid * A_ + a] + q[b * A_ + a];
        if (past) v += past[(size_t)wid * A_ + a];
        sum += tanhf(v) * attw[a];
    }
    for (int m = 16; m; m >>= 1) sum += __shfl_xor(sum, m, 32);
    if (lane == 0) s[wid] = sum + attb[0];
}

__global__ void max_kernel(const float* __restrict__ s, float* __restrict__ smax) {
    __shared__ float red[256];
    float m = -INFINITY;
    for (int i = threadIdx.x; i < NPIX; i += 256) m = fmaxf(m, s[i]);
    red[threadIdx.x] = m; __syncthreads();
    for (int k = 128; k; k >>= 1) { if (threadIdx.x < k) red[threadIdx.x] = fmaxf(red[threadIdx.x], red[threadIdx.x + k]); __syncthreads(); }
    if (threadIdx.x == 0) smax[0] = red[0];
}

__global__ void expsum_kernel(const float* __restrict__ s, const float* __restrict__ smax,
                              const float* __restrict__ mask, float* __restrict__ e,
                              float* __restrict__ esum) {
    __shared__ float red[256];
    const int b = blockIdx.x;
    const float mx = smax[0];
    float sum = 0.f;
    for (int i = threadIdx.x; i < HW_; i += 256) {
        const float ev = expf(s[b * HW_ + i] - mx) * mask[b * HW_ + i];
        e[b * HW_ + i] = ev;
        sum += ev;
    }
    red[threadIdx.x] = sum; __syncthreads();
    for (int k = 128; k; k >>= 1) { if (threadIdx.x < k) red[threadIdx.x] += red[threadIdx.x + k]; __syncthreads(); }
    if (threadIdx.x == 0) esum[b] = red[0];
}

// ct[b,d] = (sum_hw ctx[b,d,hw]*e[b,hw]) / (esum[b]+1e-10)
__global__ void ct_kernel(const float* __restrict__ ctx, const float* __restrict__ e,
                          const float* __restrict__ esum, float* __restrict__ ct) {
    const int wid = (blockIdx.x * blockDim.x + threadIdx.x) >> 5;
    const int lane = threadIdx.x & 31;
    if (wid >= B_ * D_) return;
    const int b = wid >> 9;
    const float* src = ctx + (size_t)wid * HW_;
    const float* ee = e + b * HW_;
    float sum = 0.f;
    for (int i = lane; i < HW_; i += 32) sum += src[i] * ee[i];
    for (int m = 16; m; m >>= 1) sum += __shfl_xor(sum, m, 32);
    if (lane == 0) ct[wid] = sum / (esum[b] + 1e-10f);
}

__global__ void pix_kernel(const float* __restrict__ pf, const float* __restrict__ e,
                           const float* __restrict__ esum, float* __restrict__ out) {
    const int b = blockIdx.x, k = threadIdx.x;
    if (k >= K_) return;
    const float inv = 1.f / (esum[b] + 1e-10f);
    float sum = 0.f;
    for (int i = 0; i < HW_; ++i)
        sum += e[b * HW_ + i] * pf[((size_t)b * HW_ + i) * K_ + k];
    out[b * K_ + k] = logf(sum * inv);
}

__global__ void attpast_add(float* __restrict__ ap, const float* __restrict__ e,
                            const float* __restrict__ esum) {
    const int i = blockIdx.x * blockDim.x + threadIdx.x;
    if (i < NPIX) { const int b = i >> 10; ap[i] += e[i] / (esum[b] + 1e-10f); }
}

// out[b,k] = sum_j max(o[b,2j],o[b,2j+1]) * pw[j,k] + pb[k]   (Nh=128 fixed)
__global__ void maxout_prob(const float* __restrict__ o, const float* __restrict__ pw,
                            const float* __restrict__ pb, float* __restrict__ out, int Kout) {
    __shared__ float mx[128];
    const int b = blockIdx.x, tid = threadIdx.x;  // 128 threads
    mx[tid] = fmaxf(o[b * 256 + 2 * tid], o[b * 256 + 2 * tid + 1]);
    __syncthreads();
    for (int k = tid; k < Kout; k += 128) {
        float s = pb[k];
        for (int j = 0; j < 128; ++j) s += mx[j] * pw[j * Kout + k];
        out[b * Kout + k] = s;
    }
}

// ---------------- host orchestration ----------------
extern "C" void kernel_launch(void* const* d_in, const int* in_sizes, int n_in,
                              void* d_out, int out_size, void* d_ws, size_t ws_size,
                              hipStream_t stream) {
    (void)in_sizes; (void)n_in; (void)out_size; (void)ws_size;
    const float* ctx_val    = (const float*)d_in[0];
    const float* ctx_mask   = (const float*)d_in[1];
    const int*   C_ys       = (const int*)d_in[2];
    const int*   P_ys       = (const int*)d_in[3];
    const float* P_y_masks  = (const float*)d_in[4];
    const int*   P_res      = (const int*)d_in[5];
    /* d_in[6] P_positions unused, d_in[8] length unused */
    const float* init_state = (const float*)d_in[7];
    int pi = 9;
    const float* emb_obj = (const float*)d_in[pi++];
    const float* emb_rel = (const float*)d_in[pi++];
    const float* gw[4][4];  // gru00/01/10/11 : wih, whh, bih, bhh
    for (int g = 0; g < 4; ++g)
        for (int j = 0; j < 4; ++j) gw[g][j] = (const float*)d_in[pi++];
    const float* wk_obj = (const float*)d_in[pi++];
    const float* bk_obj = (const float*)d_in[pi++];
    const float* wk_rel = (const float*)d_in[pi++];
    const float* bk_rel = (const float*)d_in[pi++];
    const float* oa_wq = (const float*)d_in[pi++];
    const float* oa_convw = (const float*)d_in[pi++];
    const float* oa_fcpast_w = (const float*)d_in[pi++];
    const float* oa_fcpast_b = (const float*)d_in[pi++];
    const float* oa_att_w = (const float*)d_in[pi++];
    const float* oa_att_b = (const float*)d_in[pi++];
    const float* ra_wq = (const float*)d_in[pi++];
    const float* ra_att_w = (const float*)d_in[pi++];
    const float* ra_att_b = (const float*)d_in[pi++];
    const float* op_ct_w = (const float*)d_in[pi++];  const float* op_ct_b = (const float*)d_in[pi++];
    const float* op_ht_w = (const float*)d_in[pi++];  const float* op_ht_b = (const float*)d_in[pi++];
    const float* op_py_w = (const float*)d_in[pi++];  const float* op_py_b = (const float*)d_in[pi++];
    const float* op_pre_w = (const float*)d_in[pi++]; const float* op_pre_b = (const float*)d_in[pi++];
    const float* op_prob_w = (const float*)d_in[pi++];const float* op_prob_b = (const float*)d_in[pi++];
    const float* rp_ct_w = (const float*)d_in[pi++];  const float* rp_ct_b = (const float*)d_in[pi++];
    const float* rp_cy_w = (const float*)d_in[pi++];  const float* rp_cy_b = (const float*)d_in[pi++];
    const float* rp_ht_w = (const float*)d_in[pi++];  const float* rp_ht_b = (const float*)d_in[pi++];
    const float* rp_prob_w = (const float*)d_in[pi++];const float* rp_prob_b = (const float*)d_in[pi++];
    const float* pix_ct_w = (const float*)d_in[pi++]; const float* pix_ct_b = (const float*)d_in[pi++];
    const float* pix_prob_w = (const float*)d_in[pi++];const float* pix_prob_b = (const float*)d_in[pi++];

    float* out = (float*)d_out;
    const size_t pc_base = 0;
    const size_t pr_base = (size_t)L_ * B_ * K_;
    const size_t pcp_base = pr_base + (size_t)L_ * B_ * KRE_;

    // workspace layout (byte-based)
    char* base = (char*)d_ws;
    size_t off = 0;
    auto allocf = [&](size_t n) { float* p = (float*)(base + off); off = (off + n * 4 + 63) & ~(size_t)63; return p; };
    auto allocb = [&](size_t n) { __bf16* p = (__bf16*)(base + off); off = (off + n * 2 + 63) & ~(size_t)63; return p; };
    float* key_obj = allocf((size_t)NPIX * A_);
    float* key_rel = allocf((size_t)NPIX * A_);
    float* past    = allocf((size_t)NPIX * A_);   // also temp for pix features (NPIX*256)
    float* pf      = allocf((size_t)NPIX * K_);
    float* w2      = allocf(128 * A_);
    float* sbuf    = allocf(NPIX);
    float* ebuf    = allocf(NPIX);
    float* esum    = allocf(B_);
    float* smax    = allocf(16);
    float* attpast = allocf(NPIX);
    float* x0      = allocf(B_ * 512);
    float* cy      = allocf(B_ * 256);
    float* qb      = allocf(B_ * 512);
    float* ctb     = allocf(B_ * 512);
    float* ct2b    = allocf(B_ * 512);
    float* ht      = allocf(B_ * 256);
    float* hthat   = allocf(B_ * 256);
    float* htr     = allocf(B_ * 256);
    float* htq     = allocf(B_ * 256);
    float* gib     = allocf(B_ * 768);
    float* ghb     = allocf(B_ * 768);
    float* obuf    = allocf(B_ * 768);

    // ---- pack weights (once per launch) ----
    auto packW = [&](const float* Wm, int Kd, int Nd) -> const __bf16* {
        __bf16* p = allocb((size_t)Kd * Nd * 2);
        const int total = (Kd / 32) * (Nd / 16) * 32;
        pack_w<<<(total + 255) / 256, 256, 0, stream>>>(Wm, Nd, Nd / 16, total, p);
        return p;
    };
    const __bf16* pk_g[4][2];
    for (int g = 0; g < 4; ++g) {
        pk_g[g][0] = packW(gw[g][0], 512, 768);   // wih
        pk_g[g][1] = packW(gw[g][1], 256, 768);   // whh
    }
    const __bf16* pk_wkobj = packW(wk_obj, 512, 512);
    const __bf16* pk_wkrel = packW(wk_rel, 512, 512);
    const __bf16* pk_pixct = packW(pix_ct_w, 512, 256);
    const __bf16* pk_oawq  = packW(oa_wq, 256, 512);
    const __bf16* pk_rawq  = packW(ra_wq, 256, 512);
    const __bf16* pk_opct  = packW(op_ct_w, 512, 256);
    const __bf16* pk_opht  = packW(op_ht_w, 256, 256);
    const __bf16* pk_oppy  = packW(op_py_w, 256, 256);
    const __bf16* pk_oppre = packW(op_pre_w, 256, 256);
    const __bf16* pk_rpct  = packW(rp_ct_w, 512, 256);
    const __bf16* pk_rpcy  = packW(rp_cy_w, 256, 256);
    const __bf16* pk_rpht  = packW(rp_ht_w, 256, 256);
    // W2 = conv (x) fcpast fusion, then pack
    w2_kernel<<<(128 * A_ + 255) / 256, 256, 0, stream>>>(oa_convw, oa_fcpast_w, w2);
    const __bf16* pk_w2 = packW(w2, 128, 512);
    // packed ctx A-operand
    __bf16* actx = allocb((size_t)NPIX * D_ * 2);
    {
        const int total = (NPIX / 16) * 16 * 32;
        pack_actx<<<(total + 255) / 256, 256, 0, stream>>>(ctx_val, actx);
    }

    auto Gx = [&](const float* X, int ldx, const __bf16* Bp, const float* bias,
                  const float* Cin, float* Out, int ldo, int M, int N, int Kd) {
        dim3 grid(N / 64, M / 16);
        gemm_xp<0><<<grid, 32, 0, stream>>>(X, ldx, Bp, N / 16, Kd / 32, bias, Cin, Out, ldo);
    };

    // ---- init + precompute ----
    init_kernel<<<(NPIX + 255) / 256, 256, 0, stream>>>(init_state, ht, htr, attpast);
    {
        dim3 gk(A_ / 64, NPIX / 16);
        gemm_pp<<<gk, 32, 0, stream>>>(actx, pk_wkobj, A_ / 16, 16, bk_obj, key_obj, A_);
        gemm_pp<<<gk, 32, 0, stream>>>(actx, pk_wkrel, A_ / 16, 16, bk_rel, key_rel, A_);
        dim3 gp(256 / 64, NPIX / 16);
        gemm_pp<<<gp, 32, 0, stream>>>(actx, pk_pixct, 256 / 16, 16, pix_ct_b, past, 256);
    }
    pix_soft<<<NPIX, 128, 0, stream>>>(past, pix_prob_w, pix_prob_b, pf);

    // ---- time loop ----
    for (int t = 0; t < L_; ++t) {
        const int* pys  = P_ys + t * B_;
        const int* pres = P_res + t * B_;
        const int* cys  = C_ys + t * B_;
        const float* mask = P_y_masks + t * B_;

        gather_kernel<<<64, 256, 0, stream>>>(emb_obj, emb_rel, pys, pres, cys, x0, cy);

        // gru00: ht_hat
        Gx(x0, 512, pk_g[0][0], gw[0][2], nullptr, gib, 768, B_, 768, 512);
        Gx(ht, 256, pk_g[0][1], gw[0][3], nullptr, ghb, 768, B_, 768, 256);
        gru_combine<<<32, 256, 0, stream>>>(gib, ghb, ht, mask, hthat);

        // coverage term: past = nbhd(att_past) @ W2 + fcpast_b
        {
            dim3 gpast(A_ / 64, NPIX / 16);
            gemm_xp<2><<<gpast, 32, 0, stream>>>(attpast, 0, pk_w2, A_ / 16, 4,
                                                 oa_fcpast_b, nullptr, past, A_);
        }
        // object attention
        Gx(hthat, 256, pk_oawq, nullptr, nullptr, qb, A_, B_, A_, 256);
        score_kernel<<<NPIX * 32 / 256, 256, 0, stream>>>(key_obj, past, qb, oa_att_w, oa_att_b, sbuf);
        max_kernel<<<1, 256, 0, stream>>>(sbuf, smax);
        expsum_kernel<<<B_, 256, 0, stream>>>(sbuf, smax, ctx_mask, ebuf, esum);
        ct_kernel<<<B_ * D_ * 32 / 256, 256, 0, stream>>>(ctx_val, ebuf, esum, ctb);
        pix_kernel<<<B_, 128, 0, stream>>>(pf, ebuf, esum, out + pcp_base + (size_t)t * B_ * K_);
        attpast_add<<<(NPIX + 255) / 256, 256, 0, stream>>>(attpast, ebuf, esum);

        // gru01: ht2 -> ht
        Gx(ctb, 512, pk_g[1][0], gw[1][2], nullptr, gib, 768, B_, 768, 512);
        Gx(hthat, 256, pk_g[1][1], gw[1][3], nullptr, ghb, 768, B_, 768, 256);
        gru_combine<<<32, 256, 0, stream>>>(gib, ghb, hthat, mask, ht);

        // object head
        Gx(ctb, 512, pk_opct, op_ct_b, nullptr, obuf, 256, B_, 256, 512);
        Gx(ht, 256, pk_opht, op_ht_b, obuf, obuf, 256, B_, 256, 256);
        Gx(x0, 512, pk_oppy, op_py_b, obuf, obuf, 256, B_, 256, 256);         // py = x0[:, :256]
        Gx(x0 + 256, 512, pk_oppre, op_pre_b, obuf, obuf, 256, B_, 256, 256); // pre = x0[:, 256:]
        maxout_prob<<<B_, 128, 0, stream>>>(obuf, op_prob_w, op_prob_b,
                                            out + pc_base + (size_t)t * B_ * K_, K_);

        // relation branch
        Gx(ctb, 512, pk_g[2][0], gw[2][2], nullptr, gib, 768, B_, 768, 512);
        Gx(htr, 256, pk_g[2][1], gw[2][3], nullptr, ghb, 768, B_, 768, 256);
        gru_combine<<<32, 256, 0, stream>>>(gib, ghb, htr, nullptr, htq);
        Gx(htq, 256, pk_rawq, nullptr, nullptr, qb, A_, B_, A_, 256);
        score_kernel<<<NPIX * 32 / 256, 256, 0, stream>>>(key_rel, nullptr, qb, ra_att_w, ra_att_b, sbuf);
        max_kernel<<<1, 256, 0, stream>>>(sbuf, smax);
        expsum_kernel<<<B_, 256, 0, stream>>>(sbuf, smax, ctx_mask, ebuf, esum);
        ct_kernel<<<B_ * D_ * 32 / 256, 256, 0, stream>>>(ctx_val, ebuf, esum, ct2b);

        Gx(ct2b, 512, pk_g[3][0], gw[3][2], nullptr, gib, 768, B_, 768, 512);
        Gx(htq, 256, pk_g[3][1], gw[3][3], nullptr, ghb, 768, B_, 768, 256);
        gru_combine<<<32, 256, 0, stream>>>(gib, ghb, htq, nullptr, htr);

        Gx(ct2b, 512, pk_rpct, rp_ct_b, nullptr, obuf, 256, B_, 256, 512);
        Gx(cy, 256, pk_rpcy, rp_cy_b, obuf, obuf, 256, B_, 256, 256);
        Gx(htr, 256, pk_rpht, rp_ht_b, obuf, obuf, 256, B_, 256, 256);
        maxout_prob<<<B_, 128, 0, stream>>>(obuf, rp_prob_w, rp_prob_b,
                                            out + pr_base + (size_t)t * B_ * KRE_, KRE_);
    }
}
